// TrueHeterogeneousRGCN_9122510537207
// MI455X (gfx1250) — matrix operand order repaired and checked by reading, not compiled, not run on hardware
//
#include <hip/hip_runtime.h>
#include <hip/hip_bf16.h>

// ---------------------------------------------------------------------------
// RGCN (basis-decomposed) on MI455X / gfx1250.
// Per layer:
//   1) Ball[i][c] : c<64 -> self_w[i][c] ; c>=64 -> W[r][i][o] from coeff*basis
//   2) one fused WMMA GEMM pass over x (A held in VGPRs, reused for all 36
//      column tiles): cols 0..63 -> agg (f32, doubles as init),
//                     cols 64..575 -> h (bf16)
//   3) edge scatter: agg[dst] += h[src, type]   (global_atomic_add_f32)
//   4) out = relu(agg + bias), ping-pong through d_out
// Fallback (ws too small for h): per-edge VALU matvec on Ball.
// ---------------------------------------------------------------------------

typedef __attribute__((ext_vector_type(16))) __bf16 v16bf;
typedef __attribute__((ext_vector_type(8)))  float  v8f;

#define DIM 64
#define RREL 8
#define RD (RREL * DIM)        // 512
#define LDB (DIM + RD)         // 576 : columns of Ball
#define NTILES (LDB / 16)      // 36

__device__ __forceinline__ unsigned short f32_to_bf16_rne(float f) {
    unsigned int u = __float_as_uint(f);
    unsigned int r = u + 0x7fffu + ((u >> 16) & 1u);
    return (unsigned short)(r >> 16);
}

// -------------------- 1) Ball = [ self_w | einsum('rb,bio->rio') ] ----------
__global__ void build_Ball_kernel(const float* __restrict__ basis_l,  // [4,64,64]
                                  const float* __restrict__ coeff_l,  // [8,4]
                                  const float* __restrict__ selfw_l,  // [64,64]
                                  float* __restrict__ Ball) {         // [64,576]
    int g = blockIdx.x * blockDim.x + threadIdx.x;
    if (g >= DIM * LDB) return;
    int i = g / LDB;
    int c = g - i * LDB;
    float acc;
    if (c < DIM) {
        acc = selfw_l[i * DIM + c];
    } else {
        int r = (c - DIM) >> 6;
        int o = (c - DIM) & (DIM - 1);
        acc = 0.f;
#pragma unroll
        for (int b = 0; b < 4; ++b)
            acc += coeff_l[r * 4 + b] * basis_l[((size_t)b * DIM + i) * DIM + o];
    }
    Ball[g] = acc;
}

// -------------------- 2) fused WMMA GEMM ------------------------------------
// One wave per 16-row strip of x.  A operands (K=64, two 16x32 bf16 blocks)
// are loaded once into VGPRs and reused across all column tiles.
// Operand packing per CDNA5 ISA 7.12.2:
//   A 16x32 bf16 : lane m=lane&15, kb8=(lane>>4)*8 ; j<8 -> k=kb8+j,
//                  j>=8 -> k=16+kb8+(j-8)
//   B 32x16 bf16 : lane n=lane&15, kb8=(lane>>4)*8 ; same k order as A
//   C 16x16 f32  : vgpr r -> row (lane>>4)*8 + r, col lane&15
__global__ void rgcn_gemm_kernel(const float* __restrict__ A,     // [M,64] fp32
                                 const float* __restrict__ Ball,  // [64,576] fp32
                                 float* __restrict__ agg,         // [M,64] f32 out
                                 unsigned short* __restrict__ h,  // [M,512] bf16 out
                                 int M, int nColTiles) {          // 36 full / 4 self-only
    int wave = blockIdx.x * (blockDim.x >> 5) + (threadIdx.x >> 5);
    int lane = threadIdx.x & 31;
    if (wave >= (M >> 4)) return;
    int row0 = wave << 4;
    int mr = lane & 15;
    int hi = lane >> 4;          // 0 or 1

    // ---- load A operands once (float4 vector loads), convert to bf16 ----
    const float* arow = A + (size_t)(row0 + mr) * DIM;
    v16bf av[2];
#pragma unroll
    for (int k0 = 0; k0 < 2; ++k0) {
        const float4* p0 = reinterpret_cast<const float4*>(arow + k0 * 32 + hi * 8);
        const float4* p1 = reinterpret_cast<const float4*>(arow + k0 * 32 + 16 + hi * 8);
        float4 f0 = p0[0], f1 = p0[1];
        float4 f2 = p1[0], f3 = p1[1];
        av[k0][0]  = (__bf16)f0.x; av[k0][1]  = (__bf16)f0.y;
        av[k0][2]  = (__bf16)f0.z; av[k0][3]  = (__bf16)f0.w;
        av[k0][4]  = (__bf16)f1.x; av[k0][5]  = (__bf16)f1.y;
        av[k0][6]  = (__bf16)f1.z; av[k0][7]  = (__bf16)f1.w;
        av[k0][8]  = (__bf16)f2.x; av[k0][9]  = (__bf16)f2.y;
        av[k0][10] = (__bf16)f2.z; av[k0][11] = (__bf16)f2.w;
        av[k0][12] = (__bf16)f3.x; av[k0][13] = (__bf16)f3.y;
        av[k0][14] = (__bf16)f3.z; av[k0][15] = (__bf16)f3.w;
    }

    // ---- sweep column tiles, A stays resident ----
    for (int tn = 0; tn < nColTiles; ++tn) {
        int col0 = tn << 4;
        v8f acc = {};
#pragma unroll
        for (int k0 = 0; k0 < 2; ++k0) {
            v16bf bv;
            const float* bcol = Ball + (size_t)(k0 * 32 + hi * 8) * LDB + col0 + mr;
#pragma unroll
            for (int j = 0; j < 16; ++j) bv[j] = (__bf16)bcol[(size_t)((j & 7) + ((j >> 3) << 4)) * LDB];
            acc = __builtin_amdgcn_wmma_f32_16x16x32_bf16(
                /*neg_a=*/false, av[k0], /*neg_b=*/false, bv,
                /*c_mod=*/(short)0, acc, /*reuse_a=*/false, /*reuse_b=*/false);
        }
        if (col0 < DIM) {        // self-loop part -> agg (f32)
#pragma unroll
            for (int r = 0; r < 8; ++r)
                agg[(size_t)(row0 + hi * 8 + r) * DIM + col0 + mr] = acc[r];
        } else {                 // relation part -> h (bf16)
#pragma unroll
            for (int r = 0; r < 8; ++r)
                h[(size_t)(row0 + hi * 8 + r) * RD + (col0 - DIM) + mr] =
                    f32_to_bf16_rne(acc[r]);
        }
    }
}

// -------------------- 3a) edge gather + scatter-add (fast path) -------------
// 32 lanes per edge; lane l handles columns 2l, 2l+1 (one dword of bf16 pair).
__global__ void edge_scatter_kernel(const unsigned short* __restrict__ h,  // [N,512] bf16
                                    const int* __restrict__ src,
                                    const int* __restrict__ dst,
                                    const int* __restrict__ etype,
                                    float* __restrict__ agg,               // [N,64]
                                    long long E) {
    long long g = (long long)blockIdx.x * blockDim.x + threadIdx.x;
    long long e = g >> 5;
    if (e >= E) return;
    int l = (int)(g & 31);
    int s = src[e], d = dst[e], t = etype[e];
    const unsigned int* hp = reinterpret_cast<const unsigned int*>(
        h + (size_t)s * RD + t * DIM + l * 2);
    unsigned int hv = *hp;
    float v0 = __uint_as_float(hv << 16);
    float v1 = __uint_as_float(hv & 0xffff0000u);
    float* ap = agg + (size_t)d * DIM + l * 2;
    unsafeAtomicAdd(ap,     v0);
    unsafeAtomicAdd(ap + 1, v1);
}

// -------------------- 3b) per-edge matvec fallback (small-ws path) ----------
__global__ void edge_matvec_kernel(const float* __restrict__ x,
                                   const int* __restrict__ src,
                                   const int* __restrict__ dst,
                                   const int* __restrict__ etype,
                                   const float* __restrict__ Ball,  // [64,576]
                                   float* __restrict__ agg,
                                   long long E) {
    long long g = (long long)blockIdx.x * blockDim.x + threadIdx.x;
    long long e = g >> 6;
    if (e >= E) return;
    int o = (int)(g & 63);
    int s = src[e], d = dst[e], t = etype[e];
    const float* xr = x + (size_t)s * DIM;
    const float* wc = Ball + DIM + t * DIM + o;   // relation block of Ball
    float acc = 0.f;
#pragma unroll
    for (int i = 0; i < DIM; ++i) acc = fmaf(xr[i], wc[(size_t)i * LDB], acc);
    unsafeAtomicAdd(agg + (size_t)d * DIM + o, acc);
}

// -------------------- 4) bias + ReLU (float4 vectorized) --------------------
__global__ void combine_relu_kernel(const float* __restrict__ agg,
                                    const float* __restrict__ bias_l,  // [64]
                                    float* __restrict__ xout, int nd4) {
    int g = blockIdx.x * blockDim.x + threadIdx.x;
    if (g >= nd4) return;
    float4 a = reinterpret_cast<const float4*>(agg)[g];
    int b0 = (g * 4) & (DIM - 1);
    float4 v;
    v.x = a.x + bias_l[b0 + 0];
    v.y = a.y + bias_l[b0 + 1];
    v.z = a.z + bias_l[b0 + 2];
    v.w = a.w + bias_l[b0 + 3];
    v.x = v.x > 0.f ? v.x : 0.f;
    v.y = v.y > 0.f ? v.y : 0.f;
    v.z = v.z > 0.f ? v.z : 0.f;
    v.w = v.w > 0.f ? v.w : 0.f;
    reinterpret_cast<float4*>(xout)[g] = v;
}

// ---------------------------------------------------------------------------
extern "C" void kernel_launch(void* const* d_in, const int* in_sizes, int n_in,
                              void* d_out, int out_size, void* d_ws, size_t ws_size,
                              hipStream_t stream) {
    const float* node_emb = (const float*)d_in[0];   // [N, 64]
    const float* basis    = (const float*)d_in[1];   // [L, 4, 64, 64]
    const float* coeff    = (const float*)d_in[2];   // [L, 8, 4]
    const float* self_w   = (const float*)d_in[3];   // [L, 64, 64]
    const float* bias     = (const float*)d_in[4];   // [L, 64]
    const int*   eidx     = (const int*)d_in[5];     // [2, E]
    const int*   etype    = (const int*)d_in[6];     // [E]

    const long long E = in_sizes[6];
    const int N = in_sizes[0] / DIM;                 // 100000 (divisible by 16)
    float* out = (float*)d_out;

    // workspace layout (256B-aligned offsets)
    const size_t BALL_BYTES = (size_t)DIM * LDB * 4;        //  144 KB
    const size_t AGG_BYTES  = (size_t)N * DIM * 4;          // 25.6 MB
    const size_t H_BYTES    = (size_t)N * RD * 2;           // 102.4 MB (bf16)
    const size_t AGG_OFF = (BALL_BYTES + 255) & ~(size_t)255;
    const size_t H_OFF   = (AGG_OFF + AGG_BYTES + 255) & ~(size_t)255;

    char* ws = (char*)d_ws;
    float*          Ball = (float*)ws;
    float*          agg  = (float*)(ws + AGG_OFF);
    unsigned short* h    = (unsigned short*)(ws + H_OFF);
    const bool full_path = ws_size >= (H_OFF + H_BYTES);

    const int* src = eidx;
    const int* dst = eidx + E;

    for (int l = 0; l < 2; ++l) {
        const float* x       = (l == 0) ? node_emb : out;
        const float* basis_l = basis  + (size_t)l * 4 * DIM * DIM;
        const float* coeff_l = coeff  + (size_t)l * RREL * 4;
        const float* selfw_l = self_w + (size_t)l * DIM * DIM;
        const float* bias_l  = bias   + (size_t)l * DIM;

        // 1) combined weight block
        build_Ball_kernel<<<(DIM * LDB + 255) / 256, 256, 0, stream>>>(
            basis_l, coeff_l, selfw_l, Ball);

        // 2) fused GEMM: agg = x@self_w (init), h = x@W (bf16)
        {
            int waves  = N / 16;                     // 6250
            int blocks = (waves + 7) / 8;            // 8 waves / 256-thread block
            rgcn_gemm_kernel<<<blocks, 256, 0, stream>>>(
                x, Ball, agg, h, N, full_path ? NTILES : 4);
        }

        if (full_path) {
            long long th = E * 32;
            edge_scatter_kernel<<<(unsigned)((th + 255) / 256), 256, 0, stream>>>(
                h, src, dst, etype, agg, E);
        } else {
            long long th = E * 64;
            edge_matvec_kernel<<<(unsigned)((th + 255) / 256), 256, 0, stream>>>(
                x, src, dst, etype, Ball, agg, E);
        }

        // 3) bias + relu -> out (ping-pong; final x == d_out contents)
        combine_relu_kernel<<<(N * DIM / 4 + 255) / 256, 256, 0, stream>>>(
            agg, bias_l, out, N * DIM / 4);
    }
}
